// RotationBlock_17746804867814
// MI455X (gfx1250) — matrix operand order repaired and checked by reading, not compiled, not run on hardware
//
#include <hip/hip_runtime.h>

#define Dv 15
#define Hv 9
#define Wv 15
#define NVOX 2025       // 15*9*15
#define NPAD 2032       // 127*16
#define CCH 256
#define K27 27
#define KTOT 6912       // 256*27
#define PLANE 1769472   // 256*6912 halves per A plane

typedef _Float16 v16h __attribute__((ext_vector_type(16)));
typedef _Float16 v8h  __attribute__((ext_vector_type(8)));
typedef float    v8f  __attribute__((ext_vector_type(8)));

// ---- output section 0: copy of x ------------------------------------------
__global__ void copy_x_kernel(const float* __restrict__ x, float* __restrict__ out) {
    int i = blockIdx.x * blockDim.x + threadIdx.x;
    if (i < CCH * NVOX) out[i] = x[i];
}

// ---- weights: fp32 (o, i, k) -> hi/lo f16 planes A[o][k*256+i] ------------
__global__ void convert_w_kernel(const float* __restrict__ w1, const float* __restrict__ w2,
                                 const float* __restrict__ w3, _Float16* __restrict__ wsA) {
    int e = blockIdx.x * blockDim.x + threadIdx.x;
    const int per = CCH * KTOT;
    if (e >= 3 * per) return;
    int t = e / per;
    int r = e - t * per;
    int o = r / KTOT;
    int q = r - o * KTOT;          // q = k*256 + i
    int k = q >> 8;
    int i = q & 255;
    const float* w = (t == 0) ? w1 : (t == 1) ? w2 : w3;
    float v = w[(size_t)o * KTOT + i * K27 + k];
    _Float16 hi = (_Float16)v;
    _Float16 lo = (_Float16)(v - (float)hi);
    size_t base = (size_t)t * 2 * PLANE + (size_t)o * KTOT + q;
    wsA[base]         = hi;
    wsA[base + PLANE] = lo;
}

// ---- gather: rotated trilinear resample -> B[n][k*256+i] hi/lo f16 --------
__global__ void gather_kernel(const float* __restrict__ x, _Float16* __restrict__ Bhi,
                              _Float16* __restrict__ Blo, int slabBase, int slabN,
                              float c, float s) {
    int tid = blockIdx.x * blockDim.x + threadIdx.x;
    if (tid >= K27 * slabN) return;
    int k = tid / slabN;
    int n = tid - k * slabN;
    size_t rowOff = (size_t)n * KTOT + (size_t)k * CCH;
    int v = slabBase + n;
    if (v >= NVOX) {                       // zero padding rows
        for (int i = 0; i < CCH; ++i) {
            Bhi[rowOff + i] = (_Float16)0.f;
            Blo[rowOff + i] = (_Float16)0.f;
        }
        return;
    }
    int d   = v / (Hv * Wv);
    int rem = v - d * (Hv * Wv);
    int h   = rem / Wv;
    int w   = rem - h * Wv;
    int kd = k / 9, r2 = k - kd * 9, kh = r2 / 3, kw = r2 - kh * 3;
    float rd = (float)(kd - 1), rh = (float)(kh - 1), rw = (float)(kw - 1);
    float pd = (float)d + rd;
    float ph = (float)h + c * rh - s * rw;
    float pw = (float)w + s * rh + c * rw;
    float bd = floorf(pd), bh = floorf(ph), bw = floorf(pw);
    float fd = pd - bd, fh = ph - bh, fw = pw - bw;
    int ibd = (int)bd, ibh = (int)bh, ibw = (int)bw;
    int   lin[8];
    float wt[8];
#pragma unroll
    for (int q = 0; q < 8; ++q) {
        int cd = (q >> 2) & 1, ch = (q >> 1) & 1, cw = q & 1;
        wt[q] = (cd ? fd : 1.f - fd) * (ch ? fh : 1.f - fh) * (cw ? fw : 1.f - fw);
        int id = min(max(ibd + cd, 0), Dv - 1);
        int ih = min(max(ibh + ch, 0), Hv - 1);
        int iw = min(max(ibw + cw, 0), Wv - 1);
        lin[q] = (id * Hv + ih) * Wv + iw;
    }
    for (int i = 0; i < CCH; ++i) {
        const float* xc = x + (size_t)i * NVOX;
        float val = wt[0] * xc[lin[0]] + wt[1] * xc[lin[1]] + wt[2] * xc[lin[2]] + wt[3] * xc[lin[3]]
                  + wt[4] * xc[lin[4]] + wt[5] * xc[lin[5]] + wt[6] * xc[lin[6]] + wt[7] * xc[lin[7]];
        _Float16 hi = (_Float16)val;
        Bhi[rowOff + i] = hi;
        Blo[rowOff + i] = (_Float16)(val - (float)hi);
    }
}

// ---- WMMA GEMM: y[o][n] = sum_kk A[o][kk]*B[n][kk], hi/lo 3-product -------
__global__ void __launch_bounds__(128)
gemm_kernel(const _Float16* __restrict__ Ahi, const _Float16* __restrict__ Alo,
            const _Float16* __restrict__ Bhi, const _Float16* __restrict__ Blo,
            const float* __restrict__ bias, float* __restrict__ out, int slabBase) {
    int lane = threadIdx.x & 31;
    int wave = threadIdx.x >> 5;
    int hlf  = lane >> 4;
    int l16  = lane & 15;
    int ntile = blockIdx.x;

    // B fragment source: lane holds column l16, K-range hlf*16..+15 (contiguous)
    const _Float16* bhiRow = Bhi + (size_t)(ntile * 16 + l16) * KTOT + hlf * 16;
    const _Float16* bloRow = Blo + (size_t)(ntile * 16 + l16) * KTOT + hlf * 16;

    // A fragment source: lane holds row o, K-range hlf*16..+15 (contiguous)
    const _Float16* ahiRow[4];
    const _Float16* aloRow[4];
#pragma unroll
    for (int mt = 0; mt < 4; ++mt) {
        int o = wave * 64 + mt * 16 + l16;
        ahiRow[mt] = Ahi + (size_t)o * KTOT + hlf * 16;
        aloRow[mt] = Alo + (size_t)o * KTOT + hlf * 16;
    }

    v8f acc[4] = {};
    for (int kb = 0; kb < KTOT; kb += 32) {
        v16h bh = *reinterpret_cast<const v16h*>(bhiRow + kb);
        v16h bl = *reinterpret_cast<const v16h*>(bloRow + kb);
#pragma unroll
        for (int mt = 0; mt < 4; ++mt) {
            v8h a0 = *reinterpret_cast<const v8h*>(ahiRow[mt] + kb);
            v8h a1 = *reinterpret_cast<const v8h*>(ahiRow[mt] + kb + 8);
            v16h ah = __builtin_shufflevector(a0, a1, 0,1,2,3,4,5,6,7,8,9,10,11,12,13,14,15);
            v8h c0 = *reinterpret_cast<const v8h*>(aloRow[mt] + kb);
            v8h c1 = *reinterpret_cast<const v8h*>(aloRow[mt] + kb + 8);
            v16h al = __builtin_shufflevector(c0, c1, 0,1,2,3,4,5,6,7,8,9,10,11,12,13,14,15);
            acc[mt] = __builtin_amdgcn_wmma_f32_16x16x32_f16(false, ah, false, bh, (short)0, acc[mt], false, false);
            acc[mt] = __builtin_amdgcn_wmma_f32_16x16x32_f16(false, ah, false, bl, (short)0, acc[mt], false, false);
            acc[mt] = __builtin_amdgcn_wmma_f32_16x16x32_f16(false, al, false, bh, (short)0, acc[mt], false, false);
        }
    }

    // C/D layout: lane l16 = column n, VGPR r -> row M = r + 8*hlf
    int nglob = slabBase + ntile * 16 + l16;
    if (nglob < NVOX) {
#pragma unroll
        for (int mt = 0; mt < 4; ++mt) {
            int obase = wave * 64 + mt * 16 + hlf * 8;
#pragma unroll
            for (int r = 0; r < 8; ++r) {
                int o = obase + r;
                out[(size_t)o * NVOX + nglob] = acc[mt][r] + bias[o];
            }
        }
    }
}

extern "C" void kernel_launch(void* const* d_in, const int* in_sizes, int n_in,
                              void* d_out, int out_size, void* d_ws, size_t ws_size,
                              hipStream_t stream) {
    const float* x  = (const float*)d_in[0];
    const float* w1 = (const float*)d_in[1];
    const float* b1 = (const float*)d_in[2];
    const float* w2 = (const float*)d_in[3];
    const float* b2 = (const float*)d_in[4];
    const float* w3 = (const float*)d_in[5];
    const float* b3 = (const float*)d_in[6];
    float*    out = (float*)d_out;
    _Float16* wsA = (_Float16*)d_ws;

    // output section 0 = x
    {
        int n = CCH * NVOX;
        copy_x_kernel<<<(n + 255) / 256, 256, 0, stream>>>(x, out);
    }
    // convert all three weight tensors to reordered hi/lo f16 planes
    {
        int n = 3 * CCH * KTOT;
        convert_w_kernel<<<(n + 255) / 256, 256, 0, stream>>>(w1, w2, w3, wsA);
    }

    // choose voxel slab size from scratch budget (deterministic in ws_size)
    size_t aBytes = (size_t)6 * PLANE * 2;  // 21,233,664 B of A planes
    int cap = 16;
    if (ws_size > aBytes) {
        size_t c = (ws_size - aBytes) / ((size_t)KTOT * 4);  // hi+lo f16 per voxel
        c = (c / 16) * 16;
        if (c > NPAD) c = NPAD;
        if (c >= 16) cap = (int)c;
    }
    _Float16* Bhi = wsA + (size_t)6 * PLANE;
    _Float16* Blo = Bhi + (size_t)cap * KTOT;

    const float* biasArr[3] = {b1, b2, b3};
    for (int t = 0; t < 3; ++t) {
        float theta = (float)(45 * (t + 1)) * (float)(3.14159265358979323846 / 180.0);
        float cth = cosf(theta), sth = sinf(theta);
        const _Float16* Ahi = wsA + (size_t)t * 2 * PLANE;
        const _Float16* Alo = Ahi + PLANE;
        float* outT = out + (size_t)(t + 1) * CCH * NVOX;
        for (int sb = 0; sb < NPAD; sb += cap) {
            int slabN = NPAD - sb;
            if (slabN > cap) slabN = cap;
            int gthreads = K27 * slabN;
            gather_kernel<<<(gthreads + 255) / 256, 256, 0, stream>>>(x, Bhi, Blo, sb, slabN, cth, sth);
            gemm_kernel<<<slabN / 16, 128, 0, stream>>>(Ahi, Alo, Bhi, Blo, biasArr[t], outT, sb);
        }
    }
}
